// MultiScaleAttention_86895778332970
// MI455X (gfx1250) — compile-verified, matrix operand order checked by reading, not run on hardware
//
#include <hip/hip_runtime.h>
#include <hip/hip_bf16.h>

// ---------------------------------------------------------------------------
// MultiScaleAttention for MI455X (gfx1250): bf16 WMMA everywhere, fp32 accum.
// B=16, D=256, H=W=64, P=8 -> patches 64, E = D*64 = 16384.
// ---------------------------------------------------------------------------

typedef __bf16 bf16;
typedef __attribute__((ext_vector_type(16))) __bf16 v16bf;
typedef __attribute__((ext_vector_type(8)))  __bf16 bf16x8;
typedef __attribute__((ext_vector_type(8)))  float  v8f;
typedef __attribute__((ext_vector_type(4)))  unsigned u32x4;
typedef __attribute__((ext_vector_type(8)))  int      i32x8;
typedef __attribute__((ext_vector_type(4)))  int      i32x4;

#define B_      16
#define D_      256
#define HW_     4096
#define E_      16384
#define NPATCH  64
#define EPS_    1e-5f

// --- TDM availability: device pass ONLY (target builtins are absent in the
// host pass, which merely needs to parse dead device bodies).  Round-4 probe
// confirmed: device pass has tensor_load_to_lds + s_wait_tensorcnt, and the
// therock TDM header is present -> 6-arg builtin form.
#if defined(__HIP_DEVICE_COMPILE__)
#if !__has_builtin(__builtin_amdgcn_tensor_load_to_lds) || \
    !__has_builtin(__builtin_amdgcn_s_wait_tensorcnt)
#error "MSA-PROBE: TDM builtins missing on DEVICE pass"
#endif
#define MSA_HAS_TDM 1
#if __has_include(<hip/amd_detail/amd_gfx1250_TDM.h>)
#define MSA_TDM_6ARG 1
#else
#define MSA_TDM_6ARG 0
#endif
#else
#define MSA_HAS_TDM 0
#endif

// ---- WMMA fragment helpers (CDNA5 16-bit A/B layout, wave32) --------------
// A (16x32): lane m = lane&15 is the row; K base kc = (lane&16)?8:0, elements
// [0..7] = K kc..kc+7, [8..15] = K 16+kc..16+kc+7 (contiguous -> 2x b128).
// B (32x16): mirrored, lane n = lane&15 is the column; all B-operands are
// stored with K contiguous so the load pattern is identical.

static __device__ inline v16bf load_frag16(const bf16* base) {
  bf16x8 lo = *(const bf16x8*)(base);
  bf16x8 hi = *(const bf16x8*)(base + 16);
  return __builtin_shufflevector(lo, hi, 0, 1, 2, 3, 4, 5, 6, 7,
                                 8, 9, 10, 11, 12, 13, 14, 15);
}

static __device__ inline void wmma_bf16(v16bf a, v16bf b, v8f& acc) {
  acc = __builtin_amdgcn_wmma_f32_16x16x32_bf16(false, a, false, b,
                                                (short)0, acc, false, false);
}

// ---------------------------------------------------------------------------
// Kernel 0: weight prep (fp32 -> bf16, conv weights re-laid [tap][o][c]),
// plus zeroing of the score accumulator (atomically reduced later).
// ---------------------------------------------------------------------------
__global__ __launch_bounds__(256)
void msa_prep(const float* __restrict__ wq, const float* __restrict__ wk,
              const float* __restrict__ wv, const float* __restrict__ w_out,
              bf16* __restrict__ wqb, bf16* __restrict__ wkb,
              bf16* __restrict__ wvb, bf16* __restrict__ wob,
              float* __restrict__ sbuf) {
  const int idx = blockIdx.x * 256 + threadIdx.x;
  if (idx < D_ * D_) {
    wqb[idx] = (bf16)wq[idx];
    wkb[idx] = (bf16)wk[idx];
    wvb[idx] = (bf16)wv[idx];
    sbuf[idx] = 0.0f;  // D*D == B*64*64 == 65536, reuse same range
  }
  if (idx < 9 * D_ * D_) {
    const int tap = idx >> 16;
    const int o = (idx >> 8) & 255;
    const int c = idx & 255;
    wob[idx] = (bf16)w_out[(o * D_ + c) * 9 + tap];
  }
}

// ---------------------------------------------------------------------------
// Kernel 1: fused Q/K/V 1x1 conv.  One WG (256 thr = 8 waves) per
// (batch, 16-spatial tile).  X tile staged via the Tensor Data Mover
// (3D tile: 8 contiguous floats x 2 rows(stride 64) x 256 ch(stride 4096)),
// then transposed/converted to bf16 [n=16][c=256] in LDS and reused across
// all 16 d-blocks x 3 weight matrices (48 WMMAs per wave).  Weight
// A-fragments are double-buffered so WMMAs overlap the global stream.
// Outputs: Q/K patch-major [b][patch][e], V transposed [b][e][patch].
// ---------------------------------------------------------------------------
__global__ __launch_bounds__(256)
void msa_qkv(const float* __restrict__ x,
             const bf16* __restrict__ wqb, const bf16* __restrict__ wkb,
             const bf16* __restrict__ wvb,
             const float* __restrict__ bq, const float* __restrict__ bk,
             const float* __restrict__ bv,
             bf16* __restrict__ qp, bf16* __restrict__ kp,
             bf16* __restrict__ vpt) {
  __shared__ __align__(16) float xf[256 * 16];  // TDM dest: [c][n] fp32
  __shared__ __align__(16) bf16 xs[16 * 256];   // GEMM operand: [n][c] bf16

  const int st = blockIdx.x;   // 0..255: (patch, s-block)
  const int b = blockIdx.y;
  const int patch = st >> 2;
  const int sb = st & 3;
  const int ph = patch >> 3, pw = patch & 7;
  const int t = threadIdx.x;

#if MSA_HAS_TDM
  // --- Tensor Data Mover: one wave issues the 3D tile descriptor ---------
  if (t < 32) {
    const int h0 = ph * 8 + sb * 2;  // two consecutive h rows of this s-block
    const int w0c = pw * 8;
    const unsigned long long ga =
        (unsigned long long)(size_t)(x + ((long)(b * D_) << 12) + (h0 << 6) + w0c);
    const unsigned ldsOff = (unsigned)(size_t)(void*)xf;  // LDS byte offset

    u32x4 g0;
    g0[0] = 1u;                                   // count=1, user desc
    g0[1] = ldsOff;                               // lds_addr
    g0[2] = (unsigned)(ga & 0xFFFFFFFFu);         // global_addr[31:0]
    g0[3] = (unsigned)((ga >> 32) & 0x01FFFFFFu)  // global_addr[56:32]
            | (2u << 30);                         // type = 2 ("image")

    i32x8 g1;
    g1[0] = (int)(2u << 16);                  // data_size = 4B
    g1[1] = (int)(8u << 16);                  // tensor_dim0 = 8 (lo16)
    g1[2] = (int)(2u << 16);                  // tensor_dim1 = 2 (lo16)
    g1[3] = (int)(8u << 16);                  // tile_dim0 = 8
    g1[4] = (int)(2u | (256u << 16));         // tile_dim1 = 2, tile_dim2 = 256
    g1[5] = 64;                               // tensor_dim0_stride = 64 (W row)
    g1[6] = (int)(0x1000u << 16);             // tensor_dim1_stride = 4096 (ch)
    g1[7] = 0;
    i32x4 g2 = {256, 0, 0, 0};                // tensor_dim2 = 256
    i32x4 g3 = {0, 0, 0, 0};

#if MSA_TDM_6ARG
    i32x8 gz = {0, 0, 0, 0, 0, 0, 0, 0};
    __builtin_amdgcn_tensor_load_to_lds(g0, g1, g2, g3, gz, 0);
#else
    __builtin_amdgcn_tensor_load_to_lds(g0, g1, g2, g3, 0);
#endif
    __builtin_amdgcn_s_wait_tensorcnt(0);
  }
#endif  // MSA_HAS_TDM
  __syncthreads();
  {  // transpose + fp32 -> bf16: xs[n][c] = xf[c][n]
    const int c = t;
#pragma unroll
    for (int n = 0; n < 16; ++n) xs[n * 256 + c] = (bf16)xf[c * 16 + n];
  }
  __syncthreads();

  const int lane = t & 31;
  const int wave = t >> 5;
  const int kc = (lane & 16) ? 8 : 0;
  const int nn = lane & 15;
  const int moff = (lane & 16) ? 8 : 0;

  const bf16* wlist[3] = {wqb, wkb, wvb};
  const float* blist[3] = {bq, bk, bv};

  for (int mb = wave; mb < 16; mb += 8) {
    const int d0 = mb * 16;
    const int arow = d0 + nn;
#pragma unroll
    for (int mat = 0; mat < 3; ++mat) {
      const bf16* wrow = wlist[mat] + arow * 256;
      const float* bias = blist[mat];
      v8f acc;
#pragma unroll
      for (int r = 0; r < 8; ++r) acc[r] = bias[d0 + r + moff];
      // double-buffered weight stream
      v16bf a_cur = load_frag16(wrow + kc);
#pragma unroll
      for (int c0 = 0; c0 < 256; c0 += 32) {
        v16bf a_nxt = a_cur;
        if (c0 < 224) {
          __builtin_prefetch(wrow + c0 + 64, 0, 1);
          a_nxt = load_frag16(wrow + c0 + 32 + kc);
        }
        v16bf bb = load_frag16(&xs[nn * 256 + c0 + kc]);
        wmma_bf16(a_cur, bb, acc);
        a_cur = a_nxt;
      }
#pragma unroll
      for (int r = 0; r < 8; ++r) {
        const int d = d0 + r + moff;
        const int e = d * 64 + sb * 16 + nn;
        const bf16 v = (bf16)acc[r];
        if (mat == 0)      qp[((long)(b * NPATCH + patch) << 14) + e] = v;
        else if (mat == 1) kp[((long)(b * NPATCH + patch) << 14) + e] = v;
        else               vpt[((((long)b << 14) + e) << 6) + patch] = v;
      }
    }
  }
}

// ---------------------------------------------------------------------------
// Kernel 2: scores = Q . K^T (M=N=64, K=16384), split 8x along K,
// fp32 atomic reduction into sbuf.  One wave per (ksplit, 16x16 tile, batch).
// Both operand streams double-buffered.
// ---------------------------------------------------------------------------
__global__ __launch_bounds__(32)
void msa_scores(const bf16* __restrict__ qp, const bf16* __restrict__ kp,
                float* __restrict__ sbuf) {
  const int ks = blockIdx.x;     // 0..7
  const int tile = blockIdx.y;   // 0..15
  const int b = blockIdx.z;
  const int mb = tile >> 2, nb = tile & 3;
  const int lane = threadIdx.x & 31;
  const int kc = (lane & 16) ? 8 : 0;
  const int nn = lane & 15;
  const int moff = (lane & 16) ? 8 : 0;

  const bf16* qrow = qp + ((long)(b * NPATCH + mb * 16 + nn) << 14);
  const bf16* krow = kp + ((long)(b * NPATCH + nb * 16 + nn) << 14);

  v8f acc;
#pragma unroll
  for (int r = 0; r < 8; ++r) acc[r] = 0.0f;

  const int e0 = ks * 2048;
  v16bf a_cur = load_frag16(qrow + e0 + kc);
  v16bf b_cur = load_frag16(krow + e0 + kc);
  for (int c0 = e0; c0 < e0 + 2048; c0 += 32) {
    v16bf a_nxt = a_cur, b_nxt = b_cur;
    if (c0 + 32 < e0 + 2048) {
      a_nxt = load_frag16(qrow + c0 + 32 + kc);
      b_nxt = load_frag16(krow + c0 + 32 + kc);
    }
    wmma_bf16(a_cur, b_cur, acc);
    a_cur = a_nxt;
    b_cur = b_nxt;
  }
#pragma unroll
  for (int r = 0; r < 8; ++r) {
    const int m = mb * 16 + r + moff;
    const int n = nb * 16 + nn;
    atomicAdd(&sbuf[(b * 64 + m) * 64 + n], acc[r]);
  }
}

// ---------------------------------------------------------------------------
// Kernel 3: softmax over 64 keys per row (scale = 1/sqrt(16384) = 1/128),
// emits bf16 attention matrix.
// ---------------------------------------------------------------------------
__global__ __launch_bounds__(64)
void msa_softmax(const float* __restrict__ sbuf, bf16* __restrict__ abuf) {
  const int b = blockIdx.x;
  const int q = threadIdx.x;
  const float* srow = sbuf + (b * 64 + q) * 64;
  bf16* arow = abuf + (b * 64 + q) * 64;
  const float scale = 0.0078125f;  // 1/128
  float s[64];
  float mx = -3.4e38f;
#pragma unroll 8
  for (int j = 0; j < 64; ++j) { s[j] = srow[j] * scale; mx = fmaxf(mx, s[j]); }
  float sum = 0.0f;
#pragma unroll 8
  for (int j = 0; j < 64; ++j) { s[j] = __expf(s[j] - mx); sum += s[j]; }
  const float inv = 1.0f / sum;
#pragma unroll 8
  for (int j = 0; j < 64; ++j) arow[j] = (bf16)(s[j] * inv);
}

// ---------------------------------------------------------------------------
// Kernel 4: out = attn . V  (M=64 patches, N=16384 embed, K=64), scattered
// back into NCHW bf16 layout for the 3x3 conv.  8 waves/WG, 1 tile/wave.
// All four fragments preloaded before the first WMMA.
// ---------------------------------------------------------------------------
__global__ __launch_bounds__(256)
void msa_attnout(const bf16* __restrict__ abuf, const bf16* __restrict__ vpt,
                 bf16* __restrict__ obuf) {
  const int b = blockIdx.y;
  const int wave = threadIdx.x >> 5;
  const int lane = threadIdx.x & 31;
  const int tileIdx = blockIdx.x * 8 + wave;  // 0..4095
  const int mb = tileIdx & 3;
  const int eb = tileIdx >> 2;                // 0..1023
  const int e0 = eb * 16;
  const int kc = (lane & 16) ? 8 : 0;
  const int nn = lane & 15;
  const int moff = (lane & 16) ? 8 : 0;

  const bf16* arow = abuf + (b * 64 + mb * 16 + nn) * 64;
  const bf16* vrow = vpt + ((((long)b << 14) + e0 + nn) << 6);

  v8f acc;
#pragma unroll
  for (int r = 0; r < 8; ++r) acc[r] = 0.0f;

  v16bf a0 = load_frag16(arow + kc);
  v16bf b0 = load_frag16(vrow + kc);
  v16bf a1 = load_frag16(arow + 32 + kc);
  v16bf b1 = load_frag16(vrow + 32 + kc);
  wmma_bf16(a0, b0, acc);
  wmma_bf16(a1, b1, acc);

#pragma unroll
  for (int r = 0; r < 8; ++r) {
    const int m = mb * 16 + r + moff;  // query patch
    const int e = e0 + nn;
    const int d = e >> 6, s = e & 63;
    const int h = (m >> 3) * 8 + (s >> 3);
    const int w = (m & 7) * 8 + (s & 7);
    obuf[((long)(b * D_ + d) << 12) + (h << 6) + w] = (bf16)acc[r];
  }
}

// ---------------------------------------------------------------------------
// Kernel 5: 3x3 conv (implicit GEMM) fused with bias + BatchNorm(eval) +
// LeakyReLU(0.2).  The 8 waves of a WG share one spatial tile: the padded
// input halo [3 rows][18 cols][256 ch] is staged once in LDS (channel-
// contiguous, zero-padding baked in), so every B-fragment is 2x ds_load_b128
// and every A-fragment 2x global b128 (double-buffered).  72 WMMAs/wave.
// ---------------------------------------------------------------------------
__global__ __launch_bounds__(256)
void msa_conv3x3(const bf16* __restrict__ obuf, const bf16* __restrict__ wob,
                 const float* __restrict__ b_out, const float* __restrict__ gamma,
                 const float* __restrict__ beta, const float* __restrict__ bn_mean,
                 const float* __restrict__ bn_var, float* __restrict__ y) {
  __shared__ __align__(16) bf16 ib[3 * 18 * 256];  // 27.6 KB halo tile

  const int b = blockIdx.y;
  const int sp = blockIdx.x >> 1;             // 0..255: (h, w-block)
  const int obbase = (blockIdx.x & 1) * 8;    // out-channel half
  const int h = sp >> 2;
  const int w0 = (sp & 3) * 16;
  const int t = threadIdx.x;

  {  // stage halo: thread t = channel c; 54 (row,col) cells, zero-padded
    const int c = t;
    const bf16* ob_b = obuf + ((long)(b * D_ + c) << 12);
    for (int rc = 0; rc < 54; ++rc) {
      const int r = rc / 18, col = rc % 18;
      const int hh = h + r - 1;
      const int ww = w0 + col - 1;
      bf16 v = (bf16)0.0f;
      if ((unsigned)hh < 64u && (unsigned)ww < 64u) v = ob_b[(hh << 6) + ww];
      ib[rc * 256 + c] = v;
    }
  }
  __syncthreads();

  const int wave = t >> 5;
  const int lane = t & 31;
  const int kc = (lane & 16) ? 8 : 0;
  const int nn = lane & 15;
  const int moff = (lane & 16) ? 8 : 0;
  const int ob = obbase + wave;
  const int orow = ob * 16 + nn;

  v8f acc;
#pragma unroll
  for (int r = 0; r < 8; ++r) acc[r] = 0.0f;

#pragma unroll
  for (int tap = 0; tap < 9; ++tap) {
    const int r = tap / 3, dw = tap % 3;
    const bf16* wtap = wob + ((tap * D_ + orow) << 8);
    const bf16* bbase = &ib[(r * 18 + nn + dw) * 256];
    v16bf a_cur = load_frag16(wtap + kc);
    for (int c0 = 0; c0 < 256; c0 += 32) {
      v16bf a_nxt = a_cur;
      if (c0 < 224) {
        __builtin_prefetch(wtap + c0 + 64, 0, 1);
        a_nxt = load_frag16(wtap + c0 + 32 + kc);
      }
      v16bf bb = load_frag16(bbase + c0 + kc);
      wmma_bf16(a_cur, bb, acc);
      a_cur = a_nxt;
    }
  }

#pragma unroll
  for (int r = 0; r < 8; ++r) {
    const int o = ob * 16 + r + moff;
    float v = acc[r] + b_out[o];
    const float inv = gamma[o] * rsqrtf(bn_var[o] + EPS_);
    v = (v - bn_mean[o]) * inv + beta[o];
    v = (v >= 0.0f) ? v : 0.2f * v;
    y[((long)(b * D_ + o) << 12) + (h << 6) + w0 + nn] = v;
  }
}

// ---------------------------------------------------------------------------
extern "C" void kernel_launch(void* const* d_in, const int* in_sizes, int n_in,
                              void* d_out, int out_size, void* d_ws, size_t ws_size,
                              hipStream_t stream) {
  const float* x = (const float*)d_in[0];
  const float* wq = (const float*)d_in[1];
  const float* bq = (const float*)d_in[2];
  const float* wk = (const float*)d_in[3];
  const float* bk = (const float*)d_in[4];
  const float* wv = (const float*)d_in[5];
  const float* bv = (const float*)d_in[6];
  const float* w_out = (const float*)d_in[7];
  const float* b_out = (const float*)d_in[8];
  const float* gamma = (const float*)d_in[9];
  const float* beta = (const float*)d_in[10];
  const float* bn_mean = (const float*)d_in[11];
  const float* bn_var = (const float*)d_in[12];
  float* y = (float*)d_out;

  // workspace carve (256-B aligned)
  char* ws = (char*)d_ws;
  size_t off = 0;
  auto alloc = [&](size_t bytes) -> void* {
    off = (off + 255) & ~(size_t)255;
    void* p = ws + off;
    off += bytes;
    return p;
  };
  bf16* qp   = (bf16*)alloc((size_t)B_ * NPATCH * E_ * 2);   // 32 MiB
  bf16* kp   = (bf16*)alloc((size_t)B_ * NPATCH * E_ * 2);   // 32 MiB
  bf16* vpt  = (bf16*)alloc((size_t)B_ * E_ * NPATCH * 2);   // 32 MiB
  bf16* obuf = (bf16*)alloc((size_t)B_ * D_ * HW_ * 2);      // 32 MiB
  float* sbuf = (float*)alloc((size_t)B_ * 64 * 64 * 4);     // 256 KiB
  bf16* abuf = (bf16*)alloc((size_t)B_ * 64 * 64 * 2);       // 128 KiB
  bf16* wqb = (bf16*)alloc((size_t)D_ * D_ * 2);
  bf16* wkb = (bf16*)alloc((size_t)D_ * D_ * 2);
  bf16* wvb = (bf16*)alloc((size_t)D_ * D_ * 2);
  bf16* wob = (bf16*)alloc((size_t)9 * D_ * D_ * 2);

  msa_prep<<<dim3((9 * D_ * D_ + 255) / 256), dim3(256), 0, stream>>>(
      wq, wk, wv, w_out, wqb, wkb, wvb, wob, sbuf);

  msa_qkv<<<dim3(256, B_), dim3(256), 0, stream>>>(
      x, wqb, wkb, wvb, bq, bk, bv, qp, kp, vpt);

  msa_scores<<<dim3(8, 16, B_), dim3(32), 0, stream>>>(qp, kp, sbuf);

  msa_softmax<<<dim3(B_), dim3(64), 0, stream>>>(sbuf, abuf);

  msa_attnout<<<dim3(512, B_), dim3(256), 0, stream>>>(abuf, vpt, obuf);

  msa_conv3x3<<<dim3(512, B_), dim3(256), 0, stream>>>(
      obuf, wob, b_out, gamma, beta, bn_mean, bn_var, y);
}
